// HeteroGNN_71554155152284
// MI455X (gfx1250) — compile-verified, matrix operand order checked by reading, not run on hardware
//
#include <hip/hip_runtime.h>
#include <hip/hip_bf16.h>

// -------------------------------------------------------------------------
// HeteroGNN for MI455X (gfx1250, wave32).
//  - All dense GEMMs run on v_wmma_f32_16x16x32_f16 (f16 in, f32 accum).
//  - A-operand tile staged to LDS once per block via CDNA5 async copies
//    (global_load_async_to_lds_b128 + s_wait_asynccnt), deduping the 8x
//    per-wave reload of the shared 16xK tile.
//  - RGCN refactored: segment_sum(h_src @ W_r) == segment_sum(h_src) @ W_r,
//    turning the per-edge GEMM into bandwidth-bound scatter + per-node GEMM.
//  - GAT: per-node score pre-reduction + 3-pass atomic segment softmax.
// -------------------------------------------------------------------------

#define GNN_N     50000
#define GNN_E     800000
#define GNN_DIN   256
#define GNN_H     128
#define GNN_R     8
#define GNN_HEADS 8
#define GNN_DH    16
#define GNN_OUT   3

typedef __attribute__((ext_vector_type(16))) _Float16 v16h;
typedef __attribute__((ext_vector_type(8)))  _Float16 v8h;
typedef __attribute__((ext_vector_type(8)))  float    v8f;

// ---------------- WMMA fragment helper (16x16x32 f16, wave32) ------------
// A/B operand: 16 rows x 32 K per fragment.
//   lanes 0-15 : row = lane,    K = [k0..k0+7] then [k0+16..k0+23]
//   lanes16-31 : row = lane-16, K = [k0+8..k0+15] then [k0+24..k0+31]
__device__ __forceinline__ v16h load_frag(const _Float16* rowbase, int ldk,
                                          int k0, int lane) {
    int row  = lane & 15;
    int koff = k0 + ((lane >> 4) << 3);   // +0 or +8
    const _Float16* p = rowbase + (size_t)row * ldk + koff;
    v8h lo = *(const v8h*)(p);
    v8h hi = *(const v8h*)(p + 16);
    v16h r;
#pragma unroll
    for (int i = 0; i < 8; ++i) { r[i] = lo[i]; r[i + 8] = hi[i]; }
    return r;
}

// ---------------- generic GEMM:  C[M x 128] = A[M x K] * Bt[128 x K] ------
// A, Bt are f16 (Bt pre-transposed: Bt[n][k] = B[k][n]).  C is f32 ld=128.
// The 16xK A tile (shared by all 8 waves of the block) is staged into LDS
// with CDNA5 async global->LDS copies; B fragments stream from global
// (weights are tiny and L2-resident).
// Launch: grid.x = M/16 blocks of 256 threads (8 waves = 8 col tiles).
#define A_LDK_PAD 8   // halves of padding per LDS row -> spreads banks

__global__ void gemm_wmma_n128(const _Float16* __restrict__ A,
                               const _Float16* __restrict__ Bt,
                               const float* __restrict__ bias,
                               const float* Cin, float* Cout,
                               _Float16* C16, int K, int do_relu) {
    __shared__ _Float16 Atile[16 * (GNN_DIN + A_LDK_PAD)];   // 8448 B max

    const int rowTile  = blockIdx.x;
    const int w        = threadIdx.x >> 5;   // wave id -> column tile
    const int lane     = threadIdx.x & 31;
    const int colBase  = w * 16;
    const int col      = lane & 15;
    const int half     = lane >> 4;          // 0: rows 0-7, 1: rows 8-15
    const int ldsK     = K + A_LDK_PAD;

    const _Float16* Arow = A  + (size_t)rowTile * 16 * K;
    const _Float16* Brow = Bt + (size_t)colBase * K;

    // ---- stage A tile: 16 rows x K halves, 16B chunks, async to LDS ----
    {
        const int chunksPerRow = K >> 3;           // 16B chunks per row
        const int totalChunks  = 16 * chunksPerRow;
        for (int c = threadIdx.x; c < totalChunks; c += blockDim.x) {
            int row = c / chunksPerRow;
            int cc  = c - row * chunksPerRow;
            const _Float16* g = Arow + (size_t)row * K + cc * 8;
            unsigned lds = (unsigned)(uintptr_t)(&Atile[row * ldsK + cc * 8]);
            asm volatile("global_load_async_to_lds_b128 %0, %1, off"
                         :: "v"(lds), "v"(g) : "memory");
        }
        asm volatile("s_wait_asynccnt 0x0" ::: "memory");
    }
    __syncthreads();

    v8f c = {};
    if (Cin) {
        const float* ci = Cin + (size_t)rowTile * 16 * GNN_H + colBase + col;
#pragma unroll
        for (int i = 0; i < 8; ++i) c[i] = ci[(size_t)(half * 8 + i) * GNN_H];
    }

    // ---- K loop: A fragments from LDS, B fragments from global ----
    {
        const int arow  = lane & 15;
        const int abase = arow * ldsK + ((lane >> 4) << 3);
        for (int k0 = 0; k0 < K; k0 += 32) {
            const _Float16* ap = &Atile[abase + k0];
            v8h alo = *(const v8h*)(ap);
            v8h ahi = *(const v8h*)(ap + 16);
            v16h a;
#pragma unroll
            for (int i = 0; i < 8; ++i) { a[i] = alo[i]; a[i + 8] = ahi[i]; }
            v16h b = load_frag(Brow, K, k0, lane);
            c = __builtin_amdgcn_wmma_f32_16x16x32_f16(
                    /*neg_a=*/false, a, /*neg_b=*/false, b,
                    /*c_mod=*/(short)0, c, /*reuse_a=*/false, /*reuse_b=*/false);
        }
    }

    if (bias) {
        float bv = bias[colBase + col];
#pragma unroll
        for (int i = 0; i < 8; ++i) c[i] += bv;
    }
    if (do_relu) {
#pragma unroll
        for (int i = 0; i < 8; ++i) c[i] = fmaxf(c[i], 0.0f);
    }

    float* co = Cout + (size_t)rowTile * 16 * GNN_H + colBase + col;
#pragma unroll
    for (int i = 0; i < 8; ++i) co[(size_t)(half * 8 + i) * GNN_H] = c[i];
    if (C16) {
        _Float16* ch = C16 + (size_t)rowTile * 16 * GNN_H + colBase + col;
#pragma unroll
        for (int i = 0; i < 8; ++i)
            ch[(size_t)(half * 8 + i) * GNN_H] = (_Float16)c[i];
    }
}

// ---------------- small utility kernels ----------------------------------
__global__ void fill_f32(float* p, float v, size_t n) {
    for (size_t i = (size_t)blockIdx.x * blockDim.x + threadIdx.x; i < n;
         i += (size_t)gridDim.x * blockDim.x)
        p[i] = v;
}

__global__ void convert_f16(const float* __restrict__ in,
                            _Float16* __restrict__ out, size_t n) {
    for (size_t i = (size_t)blockIdx.x * blockDim.x + threadIdx.x; i < n;
         i += (size_t)gridDim.x * blockDim.x)
        out[i] = (_Float16)in[i];
}

// out[b][n][k] = (f16) in[b][k][n]   (weights -> transposed f16)
__global__ void transpose_f16(const float* __restrict__ in,
                              _Float16* __restrict__ out,
                              int K, int Nc, int batch) {
    size_t total = (size_t)batch * K * Nc;
    for (size_t idx = (size_t)blockIdx.x * blockDim.x + threadIdx.x; idx < total;
         idx += (size_t)gridDim.x * blockDim.x) {
        int k = (int)(idx % K);
        int n = (int)((idx / K) % Nc);
        size_t b = idx / ((size_t)K * Nc);
        out[idx] = (_Float16)in[b * (size_t)K * Nc + (size_t)k * Nc + n];
    }
}

// ---------------- RGCN edge scatter (one wave per edge) ------------------
__global__ void rgcn_scatter(const int* __restrict__ ei,
                             const int* __restrict__ etype,
                             const float* __restrict__ h,
                             float* acc, float* deg, int E_, int rel) {
    int warp  = (int)((blockIdx.x * blockDim.x + threadIdx.x) >> 5);
    int lane  = threadIdx.x & 31;
    int nwarp = (int)((gridDim.x * blockDim.x) >> 5);
    for (int e = warp; e < E_; e += nwarp) {
        if (etype[e] != rel) continue;
        int s = ei[e], t = ei[E_ + e];
        float4 v = ((const float4*)(h + (size_t)s * GNN_H))[lane];
        float* ap = acc + (size_t)t * GNN_H + lane * 4;
        atomicAdd(ap + 0, v.x);
        atomicAdd(ap + 1, v.y);
        atomicAdd(ap + 2, v.z);
        atomicAdd(ap + 3, v.w);
        if (lane == 0) atomicAdd(deg + t, 1.0f);
    }
}

// acc16 = f16( acc / max(deg,1) )
__global__ void scale_conv(const float* __restrict__ acc,
                           const float* __restrict__ deg,
                           _Float16* __restrict__ out, size_t total) {
    for (size_t i = (size_t)blockIdx.x * blockDim.x + threadIdx.x; i < total;
         i += (size_t)gridDim.x * blockDim.x) {
        float d = deg[i >> 7];
        d = (d < 1.0f) ? 1.0f : d;
        out[i] = (_Float16)(acc[i] / d);
    }
}

// hA = relu(hB); h16 = f16(hA)
__global__ void relu_conv(const float* __restrict__ in, float* __restrict__ outf,
                          _Float16* __restrict__ outh, size_t total) {
    for (size_t i = (size_t)blockIdx.x * blockDim.x + threadIdx.x; i < total;
         i += (size_t)gridDim.x * blockDim.x) {
        float v = fmaxf(in[i], 0.0f);
        outf[i] = v;
        outh[i] = (_Float16)v;
    }
}

// ---------------- GAT ----------------------------------------------------
__global__ void gat_scores(const float* __restrict__ z,
                           const float* __restrict__ att,
                           float* __restrict__ ssrc, float* __restrict__ stgt,
                           int n_) {
    size_t total = (size_t)n_ * GNN_HEADS;
    for (size_t idx = (size_t)blockIdx.x * blockDim.x + threadIdx.x; idx < total;
         idx += (size_t)gridDim.x * blockDim.x) {
        int hd = (int)(idx & (GNN_HEADS - 1));
        size_t n = idx >> 3;
        const float* zr = z + n * GNN_H + hd * GNN_DH;
        const float* at = att + hd * 2 * GNN_DH;
        float s0 = 0.f, s1 = 0.f;
#pragma unroll
        for (int d = 0; d < GNN_DH; ++d) {
            s0 += zr[d] * at[d];
            s1 += zr[d] * at[GNN_DH + d];
        }
        ssrc[idx] = s0;
        stgt[idx] = s1;
    }
}

__device__ __forceinline__ void atomicMaxF(float* addr, float val) {
    unsigned int* u = (unsigned int*)addr;
    unsigned int old = *u;
    while (true) {
        float f = __uint_as_float(old);
        if (f >= val) break;
        unsigned int assumed = old;
        old = atomicCAS(u, assumed, __float_as_uint(val));
        if (old == assumed) break;
    }
}

__global__ void gat_alpha(const int* __restrict__ ei,
                          const float* __restrict__ ssrc,
                          const float* __restrict__ stgt,
                          float* __restrict__ alphaE, float* amax, int E_) {
    size_t total = (size_t)E_ * GNN_HEADS;
    for (size_t idx = (size_t)blockIdx.x * blockDim.x + threadIdx.x; idx < total;
         idx += (size_t)gridDim.x * blockDim.x) {
        int hd = (int)(idx & (GNN_HEADS - 1));
        size_t e = idx >> 3;
        int s = ei[e], t = ei[(size_t)E_ + e];
        float a = ssrc[(size_t)s * GNN_HEADS + hd] + stgt[(size_t)t * GNN_HEADS + hd];
        a = (a > 0.f) ? a : 0.2f * a;   // leaky relu
        alphaE[idx] = a;
        atomicMaxF(amax + (size_t)t * GNN_HEADS + hd, a);
    }
}

// One wave per edge: lane covers 4 dims, head = lane/4.
__global__ void gat_aggregate(const int* __restrict__ ei,
                              const float* __restrict__ z,
                              const float* __restrict__ alphaE,
                              const float* __restrict__ amax,
                              float* denom, float* gout, int E_) {
    int warp  = (int)((blockIdx.x * blockDim.x + threadIdx.x) >> 5);
    int lane  = threadIdx.x & 31;
    int nwarp = (int)((gridDim.x * blockDim.x) >> 5);
    int hd    = lane >> 2;
    for (int e = warp; e < E_; e += nwarp) {
        int s = ei[e], t = ei[E_ + e];
        float a  = alphaE[(size_t)e * GNN_HEADS + hd];
        float ex = __expf(a - amax[(size_t)t * GNN_HEADS + hd]);
        float4 zv = ((const float4*)(z + (size_t)s * GNN_H))[lane];
        float* op = gout + (size_t)t * GNN_H + lane * 4;
        atomicAdd(op + 0, zv.x * ex);
        atomicAdd(op + 1, zv.y * ex);
        atomicAdd(op + 2, zv.z * ex);
        atomicAdd(op + 3, zv.w * ex);
        if ((lane & 3) == 0) atomicAdd(denom + (size_t)t * GNN_HEADS + hd, ex);
    }
}

__global__ void gat_finalize(float* g, const float* __restrict__ denom,
                             const float* __restrict__ bg, size_t total) {
    for (size_t i = (size_t)blockIdx.x * blockDim.x + threadIdx.x; i < total;
         i += (size_t)gridDim.x * blockDim.x) {
        int c  = (int)(i & (GNN_H - 1));
        size_t n = i >> 7;
        float d = fmaxf(denom[n * GNN_HEADS + (c >> 4)], 1e-16f);
        g[i] = g[i] / d + bg[c];
    }
}

// ---------------- classifier + log_softmax -------------------------------
__global__ void final_kernel(const float* __restrict__ g,
                             const float* __restrict__ Wf,
                             const float* __restrict__ bf,
                             float* __restrict__ out, int n_) {
    __shared__ float wf[GNN_H * GNN_OUT];
    __shared__ float bfs[GNN_OUT];
    for (int i = threadIdx.x; i < GNN_H * GNN_OUT; i += blockDim.x) wf[i] = Wf[i];
    if (threadIdx.x < GNN_OUT) bfs[threadIdx.x] = bf[threadIdx.x];
    __syncthreads();
    int n = blockIdx.x * blockDim.x + threadIdx.x;
    if (n >= n_) return;
    float a0 = bfs[0], a1 = bfs[1], a2 = bfs[2];
    const float* gr = g + (size_t)n * GNN_H;
#pragma unroll 8
    for (int k = 0; k < GNN_H; ++k) {
        float v = gr[k];
        a0 += v * wf[k * 3 + 0];
        a1 += v * wf[k * 3 + 1];
        a2 += v * wf[k * 3 + 2];
    }
    float m = fmaxf(a0, fmaxf(a1, a2));
    float l = logf(__expf(a0 - m) + __expf(a1 - m) + __expf(a2 - m));
    out[(size_t)n * 3 + 0] = a0 - m - l;
    out[(size_t)n * 3 + 1] = a1 - m - l;
    out[(size_t)n * 3 + 2] = a2 - m - l;
}

// ---------------- host-side launch ---------------------------------------
static inline size_t align256(size_t x) { return (x + 255) & ~(size_t)255; }

extern "C" void kernel_launch(void* const* d_in, const int* in_sizes, int n_in,
                              void* d_out, int out_size, void* d_ws, size_t ws_size,
                              hipStream_t stream) {
    const float* x     = (const float*)d_in[0];
    const int*   ei    = (const int*)d_in[1];     // [2,E]
    const int*   et    = (const int*)d_in[2];     // [E]
    const float* Wp    = (const float*)d_in[3];
    const float* bp    = (const float*)d_in[4];
    const float* W1    = (const float*)d_in[5];
    const float* root1 = (const float*)d_in[6];
    const float* b1    = (const float*)d_in[7];
    const float* W2    = (const float*)d_in[8];
    const float* root2 = (const float*)d_in[9];
    const float* b2    = (const float*)d_in[10];
    const float* Wg    = (const float*)d_in[11];
    const float* att   = (const float*)d_in[12];
    const float* bg    = (const float*)d_in[13];
    const float* Wf    = (const float*)d_in[14];
    const float* bf    = (const float*)d_in[15];
    float* out = (float*)d_out;

    const int N = GNN_N, E = GNN_E, H = GNN_H, DIN = GNN_DIN, R = GNN_R;

    // ---- workspace bump allocator (~161 MB) ----
    char* ws = (char*)d_ws;
    size_t off = 0;
    auto alloc = [&](size_t bytes) -> void* {
        void* p = ws + off;
        off = align256(off + bytes);
        return p;
    };
    _Float16* x16   = (_Float16*)alloc((size_t)N * DIN * 2); // reused as alphaE later
    _Float16* WpT   = (_Float16*)alloc((size_t)DIN * H * 2);
    _Float16* r1T   = (_Float16*)alloc((size_t)H * H * 2);
    _Float16* r2T   = (_Float16*)alloc((size_t)H * H * 2);
    _Float16* W1T   = (_Float16*)alloc((size_t)R * H * H * 2);
    _Float16* W2T   = (_Float16*)alloc((size_t)R * H * H * 2);
    _Float16* WgT   = (_Float16*)alloc((size_t)H * H * 2);
    float*    hA    = (float*)alloc((size_t)N * H * 4);
    float*    hB    = (float*)alloc((size_t)N * H * 4);
    _Float16* h16   = (_Float16*)alloc((size_t)N * H * 2);
    float*    acc   = (float*)alloc((size_t)N * H * 4);
    _Float16* acc16 = (_Float16*)alloc((size_t)N * H * 2);
    float*    deg   = (float*)alloc((size_t)N * 4);
    float*    z     = (float*)alloc((size_t)N * H * 4);
    float*    ssrc  = (float*)alloc((size_t)N * GNN_HEADS * 4);
    float*    stgt  = (float*)alloc((size_t)N * GNN_HEADS * 4);
    float*    amax  = (float*)alloc((size_t)N * GNN_HEADS * 4);
    float*    denom = (float*)alloc((size_t)N * GNN_HEADS * 4);
    float*    alphaE = (float*)x16;   // overlay: E*8*4 == N*256*2 bytes
    (void)ws_size; (void)in_sizes; (void)n_in; (void)out_size;

    const int TB = 256;
    auto nb = [&](size_t n) { return (int)((n + TB - 1) / TB); };

    // ---- weight prep: transpose + f16 convert ----
    convert_f16<<<nb((size_t)N * DIN), TB, 0, stream>>>(x, x16, (size_t)N * DIN);
    transpose_f16<<<nb((size_t)DIN * H), TB, 0, stream>>>(Wp, WpT, DIN, H, 1);
    transpose_f16<<<nb((size_t)H * H), TB, 0, stream>>>(root1, r1T, H, H, 1);
    transpose_f16<<<nb((size_t)H * H), TB, 0, stream>>>(root2, r2T, H, H, 1);
    transpose_f16<<<nb((size_t)R * H * H), TB, 0, stream>>>(W1, W1T, H, H, R);
    transpose_f16<<<nb((size_t)R * H * H), TB, 0, stream>>>(W2, W2T, H, H, R);
    transpose_f16<<<nb((size_t)H * H), TB, 0, stream>>>(Wg, WgT, H, H, 1);

    const int GEMM_BLOCKS = N / 16;   // 3125
    // ---- input projection: hA = x @ Wp + bp ; h16 = f16(hA) ----
    gemm_wmma_n128<<<GEMM_BLOCKS, TB, 0, stream>>>(x16, WpT, bp, nullptr,
                                                   hA, h16, DIN, 0);

    // ---- two RGCN layers ----
    auto rgcn_layer = [&](const _Float16* WT, const _Float16* rootT, const float* b) {
        // hB = h @ root + b
        gemm_wmma_n128<<<GEMM_BLOCKS, TB, 0, stream>>>(h16, rootT, b, nullptr,
                                                       hB, nullptr, H, 0);
        for (int r = 0; r < R; ++r) {
            fill_f32<<<nb((size_t)N * H), TB, 0, stream>>>(acc, 0.0f, (size_t)N * H);
            fill_f32<<<nb((size_t)N), TB, 0, stream>>>(deg, 0.0f, (size_t)N);
            rgcn_scatter<<<6250, TB, 0, stream>>>(ei, et, hA, acc, deg, E, r);
            scale_conv<<<nb((size_t)N * H), TB, 0, stream>>>(acc, deg, acc16,
                                                             (size_t)N * H);
            // hB += (acc/deg) @ W_r
            gemm_wmma_n128<<<GEMM_BLOCKS, TB, 0, stream>>>(
                acc16, WT + (size_t)r * H * H, nullptr, hB, hB, nullptr, H, 0);
        }
        // hA = relu(hB); h16 = f16(hA)
        relu_conv<<<nb((size_t)N * H), TB, 0, stream>>>(hB, hA, h16, (size_t)N * H);
    };
    rgcn_layer(W1T, r1T, b1);
    rgcn_layer(W2T, r2T, b2);

    // ---- GAT ----
    gemm_wmma_n128<<<GEMM_BLOCKS, TB, 0, stream>>>(h16, WgT, nullptr, nullptr,
                                                   z, nullptr, H, 0);
    gat_scores<<<nb((size_t)N * GNN_HEADS), TB, 0, stream>>>(z, att, ssrc, stgt, N);
    fill_f32<<<nb((size_t)N * GNN_HEADS), TB, 0, stream>>>(amax, -1e30f,
                                                           (size_t)N * GNN_HEADS);
    gat_alpha<<<nb((size_t)E * GNN_HEADS), TB, 0, stream>>>(ei, ssrc, stgt,
                                                            alphaE, amax, E);
    fill_f32<<<nb((size_t)N * GNN_HEADS), TB, 0, stream>>>(denom, 0.0f,
                                                           (size_t)N * GNN_HEADS);
    fill_f32<<<nb((size_t)N * H), TB, 0, stream>>>(hB, 0.0f, (size_t)N * H);
    gat_aggregate<<<6250, TB, 0, stream>>>(ei, z, alphaE, amax, denom, hB, E);
    gat_finalize<<<nb((size_t)N * H), TB, 0, stream>>>(hB, denom, bg, (size_t)N * H);

    // ---- classifier + log_softmax ----
    final_kernel<<<nb((size_t)N), TB, 0, stream>>>(hB, Wf, bf, out, N);
}
